// method_8486855377176
// MI455X (gfx1250) — matrix operand order repaired and checked by reading, not compiled, not run on hardware
//
#include <hip/hip_runtime.h>
#include <math.h>

#define NN   100000
#define EE   1600000
#define FINN 256
#define HH   128
#define CCC  64
#define MPAD (HH + 8)   // padded LDS row stride (halves) -> 272B, rotates banks by 4/row

typedef __attribute__((ext_vector_type(16))) _Float16 v16h;
typedef __attribute__((ext_vector_type(8)))  float    v8f;

// ---------------- fragment loaders (ISA 7.12.2 layouts, wave32) ----------------

// A fragment 16x32 f16 from row-major f16 matrix (global or LDS after inlining).
// lane l: row = row0 + (l&15); halves 0..7 = K[k0+kh .. +7], halves 8..15 = K[k0+16+kh .. +7], kh=(l>>4)*8
__device__ __forceinline__ v16h load_a_f16(const _Float16* __restrict__ M, int ld,
                                           int row0, int k0, int lane) {
  int r  = row0 + (lane & 15);
  int kh = (lane >> 4) << 3;
  const _Float16* p = M + (size_t)r * ld + k0 + kh;
  _Float16 t[16] __attribute__((aligned(16)));
  *(uint4*)(t)     = *(const uint4*)(p);
  *(uint4*)(t + 8) = *(const uint4*)(p + 16);
  v16h a;
#pragma unroll
  for (int i = 0; i < 16; ++i) a[i] = t[i];
  return a;
}

// A fragment built from a row-major f32 matrix with per-lane (per-row) scale, converted to f16.
__device__ __forceinline__ v16h load_a_f32(const float* __restrict__ M, int ld,
                                           int row0, int k0, int lane, float scale) {
  int r  = row0 + (lane & 15);
  int kh = (lane >> 4) << 3;
  const float* p = M + (size_t)r * ld + k0 + kh;
  float t[16] __attribute__((aligned(16)));
  *(float4*)(t)      = *(const float4*)(p);
  *(float4*)(t + 4)  = *(const float4*)(p + 4);
  *(float4*)(t + 8)  = *(const float4*)(p + 16);
  *(float4*)(t + 12) = *(const float4*)(p + 20);
  v16h a;
#pragma unroll
  for (int i = 0; i < 16; ++i) a[i] = (_Float16)(t[i] * scale);
  return a;
}

// B fragment 32x16 f16 from pre-transposed weights Wt[Nout][K].
// lane l: col n = n0 + (l&15); halves 0..7 = K[k0+kh .. +7], halves 8..15 = K[k0+16+kh .. +7], kh=(l>>4)*8
__device__ __forceinline__ v16h load_b_f16(const _Float16* __restrict__ Wt, int ld,
                                           int n0, int k0, int lane) {
  int n  = n0 + (lane & 15);
  int kk = k0 + ((lane >> 4) << 3);
  const _Float16* p = Wt + (size_t)n * ld + kk;
  _Float16 t[16] __attribute__((aligned(16)));
  *(uint4*)(t)     = *(const uint4*)(p);
  *(uint4*)(t + 8) = *(const uint4*)(p + 16);
  v16h b;
#pragma unroll
  for (int i = 0; i < 16; ++i) b[i] = t[i];
  return b;
}

__device__ __forceinline__ v8f wmma_f16(v16h a, v16h b, v8f c) {
  return __builtin_amdgcn_wmma_f32_16x16x32_f16(false, a, false, b, (short)0, c, false, false);
}

__device__ __forceinline__ float sigmoidf_(float x) { return 1.0f / (1.0f + __expf(-x)); }

// ---------------- prep: W[K][N] f32 -> Wt[N][K] f16 ----------------
__global__ void transpose_convert(const float* __restrict__ W, _Float16* __restrict__ Wt,
                                  int K, int Nn) {
  int idx = blockIdx.x * blockDim.x + threadIdx.x;
  if (idx >= K * Nn) return;
  int k = idx / Nn, n = idx % Nn;
  Wt[(size_t)n * K + k] = (_Float16)W[idx];
}

// ---------------- degree ----------------
__global__ void degree_kernel(const int* __restrict__ dst, float* __restrict__ deg) {
  int e = blockIdx.x * blockDim.x + threadIdx.x;
  if (e < EE) atomicAdd(&deg[dst[e]], 1.0f);
}

__global__ void deginv_kernel(float* __restrict__ deg) {
  int i = blockIdx.x * blockDim.x + threadIdx.x;
  if (i < NN) deg[i] = 1.0f / fmaxf(deg[i], 1.0f);
}

// ---------------- encoder: h = relu(x @ enc_w + enc_b), f16 out ----------------
__global__ __launch_bounds__(256) void encoder_kernel(const float* __restrict__ x,
                                                      const _Float16* __restrict__ encT,
                                                      const float* __restrict__ enc_b,
                                                      _Float16* __restrict__ h) {
  int lane = threadIdx.x & 31;
  int tile = blockIdx.x * (blockDim.x >> 5) + (threadIdx.x >> 5);
  if (tile >= NN / 16) return;
  int row0 = tile * 16;

  v8f acc[8];
#pragma unroll
  for (int nt = 0; nt < 8; ++nt)
#pragma unroll
    for (int r = 0; r < 8; ++r) acc[nt][r] = 0.0f;

  for (int k0 = 0; k0 < FINN; k0 += 32) {
    v16h a = load_a_f32(x, FINN, row0, k0, lane, 1.0f);
#pragma unroll
    for (int nt = 0; nt < 8; ++nt) {
      v16h b = load_b_f16(encT, FINN, nt * 16, k0, lane);
      acc[nt] = wmma_f16(a, b, acc[nt]);
    }
  }

  int cl = lane & 15;
  int rb = row0 + ((lane >> 4) << 3);
#pragma unroll
  for (int nt = 0; nt < 8; ++nt) {
    int col = nt * 16 + cl;
    float bia = enc_b[col];
#pragma unroll
    for (int r = 0; r < 8; ++r) {
      float v = acc[nt][r] + bia;
      v = v > 0.0f ? v : 0.0f;
      h[(size_t)(rb + r) * HH + col] = (_Float16)v;
    }
  }
}

// ---------------- edge aggregate: agg[dst] += h[src] (f32 atomics, f16 gather) ----------------
__global__ __launch_bounds__(256) void aggregate_kernel(const int* __restrict__ src,
                                                        const int* __restrict__ dst,
                                                        const _Float16* __restrict__ h,
                                                        float* __restrict__ agg) {
  unsigned int gid = blockIdx.x * 256u + threadIdx.x;
  unsigned int e = gid >> 4;               // 16 chunks of 8 features per edge
  if (e >= EE) return;
  int c = (int)(gid & 15u) * 8;
  int s = src[e], d = dst[e];
  _Float16 t[8] __attribute__((aligned(16)));
  *(uint4*)t = *(const uint4*)(h + (size_t)s * HH + c);
  float* ap = agg + (size_t)d * HH + c;
#pragma unroll
  for (int i = 0; i < 8; ++i) atomicAdd(ap + i, (float)t[i]);
}

// ---------------- fused conv+gate, m tile staged in LDS ----------------
// m = h@Wself + (agg*deginv)@Wneigh + cb         (WMMA, D-layout -> LDS tile)
// h_out = sig(h@Wh1+m@Wm1+b1)*h + sig(h@Wh2+m@Wm2+b2)*m   (m A-fragments via ds_load_b128)
__global__ __launch_bounds__(256) void conv_gate_kernel(const _Float16* __restrict__ h,
                                                        const float* __restrict__ agg,
                                                        const float* __restrict__ deginv,
                                                        const _Float16* __restrict__ wsT,
                                                        const _Float16* __restrict__ wnT,
                                                        const float* __restrict__ cb,
                                                        const _Float16* __restrict__ wh1T,
                                                        const _Float16* __restrict__ wm1T,
                                                        const float* __restrict__ b1,
                                                        const _Float16* __restrict__ wh2T,
                                                        const _Float16* __restrict__ wm2T,
                                                        const float* __restrict__ b2,
                                                        _Float16* __restrict__ hout) {
  __shared__ _Float16 msh[128 * MPAD];     // 34,816 B: 128-row m tile, padded rows

  int lane  = threadIdx.x & 31;
  int wave  = threadIdx.x >> 5;
  int tile  = blockIdx.x * (blockDim.x >> 5) + wave;
  bool live = tile < NN / 16;
  int row0  = tile * 16;                   // global row base of this wave's stripe
  int lrow0 = wave * 16;                   // LDS-local row base
  int cl    = lane & 15;
  int rsub  = (lane >> 4) << 3;            // 0 or 8 (D-layout row offset)

  // ---- phase 1: conv -> m tile in LDS ----
  if (live) {
    float dinv = deginv[row0 + cl];        // per-lane A-row scale

    v8f acc[8];
#pragma unroll
    for (int nt = 0; nt < 8; ++nt)
#pragma unroll
      for (int r = 0; r < 8; ++r) acc[nt][r] = 0.0f;

    for (int k0 = 0; k0 < HH; k0 += 32) {
      v16h ah = load_a_f16(h, HH, row0, k0, lane);
      v16h ag = load_a_f32(agg, HH, row0, k0, lane, dinv);
#pragma unroll
      for (int nt = 0; nt < 8; ++nt) {
        acc[nt] = wmma_f16(ah, load_b_f16(wsT, HH, nt * 16, k0, lane), acc[nt]);
        acc[nt] = wmma_f16(ag, load_b_f16(wnT, HH, nt * 16, k0, lane), acc[nt]);
      }
    }

#pragma unroll
    for (int nt = 0; nt < 8; ++nt) {
      int col = nt * 16 + cl;
      float bia = cb[col];
#pragma unroll
      for (int r = 0; r < 8; ++r)
        msh[(lrow0 + rsub + r) * MPAD + col] = (_Float16)(acc[nt][r] + bia);
    }
  }

  __syncthreads();

  // ---- phase 2: gate (m read from LDS) ----
  if (live) {
    v8f acca[8], accb[8];
#pragma unroll
    for (int nt = 0; nt < 8; ++nt)
#pragma unroll
      for (int r = 0; r < 8; ++r) { acca[nt][r] = 0.0f; accb[nt][r] = 0.0f; }

    for (int k0 = 0; k0 < HH; k0 += 32) {
      v16h ah = load_a_f16(h, HH, row0, k0, lane);
      // m A-fragment from LDS (ds_load_b128 x2 per lane, conflict-free via MPAD)
      v16h am;
      {
        int kh = (lane >> 4) << 3;
        const _Float16* p = &msh[(lrow0 + cl) * MPAD + k0 + kh];
        _Float16 t[16] __attribute__((aligned(16)));
        *(uint4*)(t)     = *(const uint4*)(p);
        *(uint4*)(t + 8) = *(const uint4*)(p + 16);
#pragma unroll
        for (int i = 0; i < 16; ++i) am[i] = t[i];
      }
#pragma unroll
      for (int nt = 0; nt < 8; ++nt) {
        acca[nt] = wmma_f16(ah, load_b_f16(wh1T, HH, nt * 16, k0, lane), acca[nt]);
        acca[nt] = wmma_f16(am, load_b_f16(wm1T, HH, nt * 16, k0, lane), acca[nt]);
        accb[nt] = wmma_f16(ah, load_b_f16(wh2T, HH, nt * 16, k0, lane), accb[nt]);
        accb[nt] = wmma_f16(am, load_b_f16(wm2T, HH, nt * 16, k0, lane), accb[nt]);
      }
    }

#pragma unroll
    for (int nt = 0; nt < 8; ++nt) {
      int col = nt * 16 + cl;
      float bb1 = b1[col], bb2 = b2[col];
#pragma unroll
      for (int r = 0; r < 8; ++r) {
        int row = row0 + rsub + r;
        float alpha = sigmoidf_(acca[nt][r] + bb1);
        float beta  = sigmoidf_(accb[nt][r] + bb2);
        float hv = (float)h[(size_t)row * HH + col];
        float mv = (float)msh[(lrow0 + rsub + r) * MPAD + col];
        hout[(size_t)row * HH + col] = (_Float16)(alpha * hv + beta * mv);
      }
    }
  }
}

// ---------------- decoder: out = h @ dec_w + dec_b (f32 out) ----------------
__global__ __launch_bounds__(256) void decoder_kernel(const _Float16* __restrict__ h,
                                                      const _Float16* __restrict__ decT,
                                                      const float* __restrict__ dec_b,
                                                      float* __restrict__ out) {
  int lane = threadIdx.x & 31;
  int tile = blockIdx.x * (blockDim.x >> 5) + (threadIdx.x >> 5);
  if (tile >= NN / 16) return;
  int row0 = tile * 16;

  v8f acc[4];
#pragma unroll
  for (int nt = 0; nt < 4; ++nt)
#pragma unroll
    for (int r = 0; r < 8; ++r) acc[nt][r] = 0.0f;

  for (int k0 = 0; k0 < HH; k0 += 32) {
    v16h a = load_a_f16(h, HH, row0, k0, lane);
#pragma unroll
    for (int nt = 0; nt < 4; ++nt) {
      v16h b = load_b_f16(decT, HH, nt * 16, k0, lane);
      acc[nt] = wmma_f16(a, b, acc[nt]);
    }
  }

  int cl = lane & 15;
  int rb = row0 + ((lane >> 4) << 3);
#pragma unroll
  for (int nt = 0; nt < 4; ++nt) {
    int col = nt * 16 + cl;
    float bia = dec_b[col];
#pragma unroll
    for (int r = 0; r < 8; ++r)
      out[(size_t)(rb + r) * CCC + col] = acc[nt][r] + bia;
  }
}

// ---------------- host orchestration ----------------
extern "C" void kernel_launch(void* const* d_in, const int* in_sizes, int n_in,
                              void* d_out, int out_size, void* d_ws, size_t ws_size,
                              hipStream_t stream) {
  const float* x     = (const float*)d_in[0];
  const int*   ei    = (const int*)d_in[1];
  const float* enc_w = (const float*)d_in[2];
  const float* enc_b = (const float*)d_in[3];
  const float* cws   = (const float*)d_in[4];   // [L,H,H]
  const float* cwn   = (const float*)d_in[5];   // [L,H,H]
  const float* cb    = (const float*)d_in[6];   // [L,H]
  const float* gwh1  = (const float*)d_in[7];
  const float* gwm1  = (const float*)d_in[8];
  const float* gb1   = (const float*)d_in[9];
  const float* gwh2  = (const float*)d_in[10];
  const float* gwm2  = (const float*)d_in[11];
  const float* gb2   = (const float*)d_in[12];
  const float* dec_w = (const float*)d_in[13];
  const float* dec_b = (const float*)d_in[14];
  const int* src = ei;
  const int* dst = ei + EE;
  float* out = (float*)d_out;

  // workspace partition (256B aligned)
  char* w = (char*)d_ws;
  auto alloc = [&](size_t bytes) -> char* {
    char* p = w; w += (bytes + 255) & ~(size_t)255; return p;
  };
  float*    deg   = (float*)alloc((size_t)NN * sizeof(float));
  float*    agg   = (float*)alloc((size_t)NN * HH * sizeof(float));
  _Float16* h0    = (_Float16*)alloc((size_t)NN * HH * 2);
  _Float16* h1    = (_Float16*)alloc((size_t)NN * HH * 2);
  _Float16* encT  = (_Float16*)alloc((size_t)FINN * HH * 2);
  _Float16* cwsT0 = (_Float16*)alloc((size_t)HH * HH * 2);
  _Float16* cwsT1 = (_Float16*)alloc((size_t)HH * HH * 2);
  _Float16* cwnT0 = (_Float16*)alloc((size_t)HH * HH * 2);
  _Float16* cwnT1 = (_Float16*)alloc((size_t)HH * HH * 2);
  _Float16* wh1T  = (_Float16*)alloc((size_t)HH * HH * 2);
  _Float16* wm1T  = (_Float16*)alloc((size_t)HH * HH * 2);
  _Float16* wh2T  = (_Float16*)alloc((size_t)HH * HH * 2);
  _Float16* wm2T  = (_Float16*)alloc((size_t)HH * HH * 2);
  _Float16* decT  = (_Float16*)alloc((size_t)HH * CCC * 2);

  dim3 blk(256);
  const int gemmGrid = (NN / 16 + 7) / 8;   // 8 waves per block, 16 rows per wave

  // weights -> f16 transposed
  auto tc = [&](const float* W, _Float16* Wt, int K, int Nn) {
    int tot = K * Nn;
    transpose_convert<<<dim3((tot + 255) / 256), blk, 0, stream>>>(W, Wt, K, Nn);
  };
  tc(enc_w, encT, FINN, HH);
  tc(cws,           cwsT0, HH, HH);
  tc(cws + HH * HH, cwsT1, HH, HH);
  tc(cwn,           cwnT0, HH, HH);
  tc(cwn + HH * HH, cwnT1, HH, HH);
  tc(gwh1, wh1T, HH, HH);
  tc(gwm1, wm1T, HH, HH);
  tc(gwh2, wh2T, HH, HH);
  tc(gwm2, wm2T, HH, HH);
  tc(dec_w, decT, HH, CCC);

  // degree -> deg_inv (in place)
  hipMemsetAsync(deg, 0, (size_t)NN * sizeof(float), stream);
  degree_kernel<<<dim3((EE + 255) / 256), blk, 0, stream>>>(dst, deg);
  deginv_kernel<<<dim3((NN + 255) / 256), blk, 0, stream>>>(deg);

  // encoder
  encoder_kernel<<<dim3(gemmGrid), blk, 0, stream>>>(x, encT, enc_b, h0);

  // layer 0: h0 -> h1
  hipMemsetAsync(agg, 0, (size_t)NN * HH * sizeof(float), stream);
  aggregate_kernel<<<dim3((unsigned)((size_t)EE * 16 / 256)), blk, 0, stream>>>(src, dst, h0, agg);
  conv_gate_kernel<<<dim3(gemmGrid), blk, 0, stream>>>(h0, agg, deg, cwsT0, cwnT0, cb,
                                                       wh1T, wm1T, gb1, wh2T, wm2T, gb2, h1);

  // layer 1: h1 -> h0
  hipMemsetAsync(agg, 0, (size_t)NN * HH * sizeof(float), stream);
  aggregate_kernel<<<dim3((unsigned)((size_t)EE * 16 / 256)), blk, 0, stream>>>(src, dst, h1, agg);
  conv_gate_kernel<<<dim3(gemmGrid), blk, 0, stream>>>(h1, agg, deg, cwsT1, cwnT1, cb + HH,
                                                       wh1T, wm1T, gb1, wh2T, wm2T, gb2, h0);

  // decoder
  decoder_kernel<<<dim3(gemmGrid), blk, 0, stream>>>(h0, decT, dec_b, out);
}
